// Generator_89163521065392
// MI455X (gfx1250) — matrix operand order repaired and checked by reading, not compiled, not run on hardware
//
#include <hip/hip_runtime.h>
#include <hip/hip_bf16.h>
#include <math.h>

typedef __attribute__((ext_vector_type(16))) __bf16 v16bf;
typedef __attribute__((ext_vector_type(8)))  __bf16 v8bf;
typedef __attribute__((ext_vector_type(8)))  float  v8f;

typedef __attribute__((ext_vector_type(4))) unsigned int tdm_u32x4;
typedef __attribute__((ext_vector_type(8))) int          tdm_i32x8;
typedef __attribute__((ext_vector_type(4))) int          tdm_i32x4;

#define VOCAB   8192
#define EMBED   256
#define HIDDEN  1024
#define LATENT  128
#define MAX_LEN 128
#define BATCH   128
#define SPLITK  8

// ---------------------------------------------------------------------------
// One-time preprocessing kernels
// ---------------------------------------------------------------------------
__global__ void k_f32_to_bf16(const float* __restrict__ in, __bf16* __restrict__ out, int n) {
  int i = blockIdx.x * blockDim.x + threadIdx.x;
  if (i < n) out[i] = (__bf16)in[i];
}

// E [VOCAB, EMBED] f32 row-major  ->  Et [EMBED, VOCAB] bf16 row-major ([N,K] form)
__global__ void k_transpose_E(const float* __restrict__ E, __bf16* __restrict__ Et) {
  int i = blockIdx.x * blockDim.x + threadIdx.x;
  if (i < VOCAB * EMBED) {
    int v = i / EMBED, d = i % EMBED;
    Et[(size_t)d * VOCAB + v] = (__bf16)E[i];
  }
}

// h0 = noise @ W_init.T + b_init   (tiny: 16.7M MACs, plain fp32)
__global__ void k_h0(const float* __restrict__ noise, const float* __restrict__ Wi,
                     const float* __restrict__ bi, float* __restrict__ h,
                     __bf16* __restrict__ hbf) {
  int i = blockIdx.x * blockDim.x + threadIdx.x;
  if (i >= BATCH * HIDDEN) return;
  int m = i >> 10, n = i & (HIDDEN - 1);
  const float* nr = noise + (size_t)m * LATENT;
  const float* wr = Wi + (size_t)n * LATENT;
  float acc = bi[n];
  for (int k = 0; k < LATENT; ++k) acc += nr[k] * wr[k];
  h[i] = acc;
  hbf[i] = (__bf16)acc;
}

// y_prev for step 0 = one-hot(PAD_IDX=0) per row
__global__ void k_init_y(__bf16* __restrict__ y) {
  int i = blockIdx.x * blockDim.x + threadIdx.x;
  if (i < BATCH * VOCAB) y[i] = (__bf16)(((i & (VOCAB - 1)) == 0) ? 1.0f : 0.0f);
}

// ---------------------------------------------------------------------------
// Shared fragment helpers (CDNA5 16-bit A(16x32)/B(32x16) VGPR layouts)
// ---------------------------------------------------------------------------
__device__ __forceinline__ v16bf load_a_frag(const __bf16* arow, int kk, int half) {
  v8bf alo = *(const v8bf*)(arow + kk + half * 16);
  v8bf ahi = *(const v8bf*)(arow + kk + half * 16 + 8);
  v16bf a;
#pragma unroll
  for (int i = 0; i < 8; ++i) { a[i] = alo[i]; a[8 + i] = ahi[i]; }
  return a;
}

// ---------------------------------------------------------------------------
// TDM-staged bf16 WMMA GEMM:  C[128,N] = A[128,K] @ B[N,K]^T (+bias)
// One block = 4 waves sharing a 16-row A tile (16 x K bf16, <=32KB LDS),
// DMA'd into LDS by the Tensor Data Mover.  Each wave owns a 16x64 C tile:
// A frag from LDS (ds_load_b128), 4 B frags from global (global_load_b128),
// 4 v_wmma per K-chunk.
// ---------------------------------------------------------------------------
__global__ __launch_bounds__(128) void k_gemm_tdm(
    const __bf16* __restrict__ A, const __bf16* __restrict__ B,
    const float* __restrict__ bias, float* __restrict__ Cf,
    __bf16* __restrict__ Cb, int K, int N) {
  __shared__ __bf16 As[16 * 1024];           // max K = 1024 here

  const int mtile = blockIdx.x & 7;          // M = 128 -> 8 m-tiles
  const int ng    = blockIdx.x >> 3;         // group of 4 n-quads (256 cols)
  const int wv    = threadIdx.x >> 5;
  const int lane  = threadIdx.x & 31;
  const int half  = lane >> 4;
  const int l16   = lane & 15;

  // --- TDM: DMA A tile [16, K] bf16 into LDS (wave 0 issues & waits) ---
  if (wv == 0) {
    unsigned lds = (unsigned)(uintptr_t)(&As[0]);      // low 32b of generic = LDS offset
    unsigned long long ga =
        (unsigned long long)(uintptr_t)(A + (size_t)mtile * 16 * K);
    tdm_u32x4 g0 = { 1u,                                // count=1, no gather
                     lds,                               // lds_addr
                     (unsigned)(ga & 0xFFFFFFFFu),
                     (unsigned)((ga >> 32) & 0x01FFFFFFu) | (2u << 30) };  // type=2
    unsigned uK = (unsigned)K;
    tdm_i32x8 g1 = { (int)0x00010000u,                  // data_size=1 (2 bytes)
                     (int)((uK & 0xFFFFu) << 16),       // tensor_dim0[15:0]
                     (int)((uK >> 16) | (128u << 16)),  // tensor_dim0[31:16] | tensor_dim1[15:0]
                     (int)(uK << 16),                   // tensor_dim1[31:16]=0 | tile_dim0=K
                     16,                                // tile_dim1=16, tile_dim2=0
                     (int)uK,                           // tensor_dim0_stride[31:0]
                     0, 0 };
    tdm_i32x4 g2 = { 0, 0, 0, 0 };
    tdm_i32x4 g3 = { 0, 0, 0, 0 };
#if defined(__clang_major__) && (__clang_major__ >= 23)
    tdm_i32x8 g4 = { 0, 0, 0, 0, 0, 0, 0, 0 };
    __builtin_amdgcn_tensor_load_to_lds(g0, g1, g2, g3, g4, 0);
#else
    __builtin_amdgcn_tensor_load_to_lds(g0, g1, g2, g3, 0);
#endif
    __builtin_amdgcn_s_wait_tensorcnt(0);
  }
  __syncthreads();

  const int nt4 = (ng << 2) + wv;            // this wave's 64-col group
  const int n0  = (nt4 << 6) + l16;
  const __bf16* brow[4];
#pragma unroll
  for (int j = 0; j < 4; ++j) brow[j] = B + (size_t)(n0 + 16 * j) * K;

  const __bf16* arow = &As[(size_t)l16 * K];

  v8f acc[4];
#pragma unroll
  for (int j = 0; j < 4; ++j) acc[j] = (v8f){};

  for (int kk = 0; kk < K; kk += 32) {
    v16bf a = load_a_frag(arow, kk, half);   // ds_load_b128 x2
    const int kb = kk + half * 16;
    v16bf bf[4];
#pragma unroll
    for (int j = 0; j < 4; ++j) bf[j] = *(const v16bf*)(brow[j] + kb);
    if (kk + 32 < K) {                       // keep next B chunk near (WGP scope)
      __builtin_prefetch(brow[0] + kb + 32, 0, 3);
      __builtin_prefetch(brow[2] + kb + 32, 0, 3);
    }
#pragma unroll
    for (int j = 0; j < 4; ++j)
      acc[j] = __builtin_amdgcn_wmma_f32_16x16x32_bf16(
          false, a, false, bf[j], (short)0, acc[j], false, false);
  }

#pragma unroll
  for (int j = 0; j < 4; ++j) {
    const int n = n0 + 16 * j;
    const float bb = bias ? bias[n] : 0.0f;
#pragma unroll
    for (int v = 0; v < 8; ++v) {
      const int m = (mtile << 4) + (half << 3) + v;
      const float val = acc[j][v] + bb;
      if (Cf) Cf[(size_t)m * N + n] = val;
      if (Cb) Cb[(size_t)m * N + n] = (__bf16)val;
    }
  }
}

// ---------------------------------------------------------------------------
// Split-K WMMA GEMM for x = y_prev @ E  (M=128, N=256, K=8192 -> 8 splits)
// Writes f32 partials; deterministic fixed-order reduction follows.
// ---------------------------------------------------------------------------
__global__ __launch_bounds__(128) void k_gemm_x_splitk(
    const __bf16* __restrict__ A,   // y_bf [128, 8192]
    const __bf16* __restrict__ B,   // Et   [256, 8192]
    float* __restrict__ Cpart) {    // [SPLITK][128][256]
  const int lane = threadIdx.x & 31;
  const int wave = blockIdx.x * (blockDim.x >> 5) + (threadIdx.x >> 5);
  if (wave >= SPLITK * 32) return;
  const int split = wave >> 5;               // 0..7
  const int w     = wave & 31;               // 8 mtiles x 4 nquads
  const int mtile = w & 7;
  const int nt4   = w >> 3;

  const int half = lane >> 4;
  const int l16  = lane & 15;
  const int mrow = (mtile << 4) + l16;
  const int K = VOCAB;

  const __bf16* arow = A + (size_t)mrow * K;
  const int n0 = (nt4 << 6) + l16;
  const __bf16* brow[4];
#pragma unroll
  for (int j = 0; j < 4; ++j) brow[j] = B + (size_t)(n0 + 16 * j) * K;

  v8f acc[4];
#pragma unroll
  for (int j = 0; j < 4; ++j) acc[j] = (v8f){};

  const int k0 = split * (K / SPLITK), k1 = k0 + K / SPLITK;
  for (int kk = k0; kk < k1; kk += 32) {
    v16bf a = load_a_frag(arow, kk, half);
    const int kb = kk + half * 16;
    v16bf bf[4];
#pragma unroll
    for (int j = 0; j < 4; ++j) bf[j] = *(const v16bf*)(brow[j] + kb);
    if (kk + 32 < k1) {
      __builtin_prefetch(arow + kk + 32, 0, 3);
      __builtin_prefetch(brow[0] + kb + 32, 0, 3);
      __builtin_prefetch(brow[2] + kb + 32, 0, 3);
    }
#pragma unroll
    for (int j = 0; j < 4; ++j)
      acc[j] = __builtin_amdgcn_wmma_f32_16x16x32_bf16(
          false, a, false, bf[j], (short)0, acc[j], false, false);
  }

  float* Cs = Cpart + (size_t)split * BATCH * EMBED;
#pragma unroll
  for (int j = 0; j < 4; ++j) {
    const int n = n0 + 16 * j;
#pragma unroll
    for (int v = 0; v < 8; ++v) {
      const int m = (mtile << 4) + (half << 3) + v;
      Cs[(size_t)m * EMBED + n] = acc[j][v];
    }
  }
}

__global__ void k_reduce_x(const float* __restrict__ Cpart, __bf16* __restrict__ xbf) {
  int i = blockIdx.x * blockDim.x + threadIdx.x;
  if (i >= BATCH * EMBED) return;
  float s = 0.0f;
#pragma unroll
  for (int j = 0; j < SPLITK; ++j) s += Cpart[(size_t)j * BATCH * EMBED + i];
  xbf[i] = (__bf16)s;
}

// ---------------------------------------------------------------------------
// GRU gate fusion: h = (1-z)*n + z*h  (gate order r,z,n)
// ---------------------------------------------------------------------------
__global__ void k_gru(const float* __restrict__ gi, const float* __restrict__ gh,
                      float* __restrict__ h, __bf16* __restrict__ hbf) {
  int i = blockIdx.x * blockDim.x + threadIdx.x;
  if (i >= BATCH * HIDDEN) return;
  int m = i >> 10, n = i & (HIDDEN - 1);
  const float* gim = gi + (size_t)m * 3 * HIDDEN;
  const float* ghm = gh + (size_t)m * 3 * HIDDEN;
  float r = 1.0f / (1.0f + __expf(-(gim[n] + ghm[n])));
  float z = 1.0f / (1.0f + __expf(-(gim[HIDDEN + n] + ghm[HIDDEN + n])));
  float nn = tanhf(gim[2 * HIDDEN + n] + r * ghm[2 * HIDDEN + n]);
  float hv = (1.0f - z) * nn + z * h[i];
  h[i] = hv;
  hbf[i] = (__bf16)hv;
}

// ---------------------------------------------------------------------------
// Gumbel-softmax over V=8192 per row; threefry2x32 counter-based noise
// ---------------------------------------------------------------------------
__device__ __forceinline__ unsigned rotl32(unsigned x, int r) {
  return (x << r) | (x >> (32 - r));
}
__device__ __forceinline__ void threefry2x32(unsigned k0, unsigned k1,
                                             unsigned c0, unsigned c1,
                                             unsigned& o0, unsigned& o1) {
  unsigned ks[3] = {k0, k1, k0 ^ k1 ^ 0x1BD11BDAu};
  unsigned x0 = c0 + k0, x1 = c1 + k1;
  const int R0[4] = {13, 15, 26, 6}, R1[4] = {17, 29, 16, 24};
#pragma unroll
  for (int g = 0; g < 5; ++g) {
    const int* R = (g & 1) ? R1 : R0;
#pragma unroll
    for (int r = 0; r < 4; ++r) { x0 += x1; x1 = rotl32(x1, R[r]); x1 ^= x0; }
    x0 += ks[(g + 1) % 3];
    x1 += ks[(g + 2) % 3] + (unsigned)(g + 1);
  }
  o0 = x0; o1 = x1;
}
__device__ __forceinline__ float gumbelf(int t, int m, int v) {
  unsigned o0, o1;
  threefry2x32(0x2Au, (unsigned)t, (unsigned)m, (unsigned)v, o0, o1);
  float u = __uint_as_float((o0 >> 9) | 0x3F800000u) - 1.0f;  // [0,1)
  u = fminf(fmaxf(u, 1e-12f), 1.0f - 1e-7f);
  return -__logf(-__logf(u));
}

__global__ __launch_bounds__(256) void k_softmax(
    const float* __restrict__ logits, const float* __restrict__ temperature,
    int t, float* __restrict__ out, __bf16* __restrict__ ybf) {
  __shared__ float red[256];
  const int m = blockIdx.x, tid = threadIdx.x;
  const float inv_tau = 1.0f / temperature[0];
  const float* lrow = logits + (size_t)m * VOCAB;

  float av[32];
  float mx = -INFINITY;
#pragma unroll
  for (int j = 0; j < 32; ++j) {
    int v = tid + j * 256;              // coalesced
    float a = (lrow[v] + gumbelf(t, m, v)) * inv_tau;
    av[j] = a;
    mx = fmaxf(mx, a);
  }
  red[tid] = mx; __syncthreads();
  for (int s = 128; s > 0; s >>= 1) {
    if (tid < s) red[tid] = fmaxf(red[tid], red[tid + s]);
    __syncthreads();
  }
  mx = red[0]; __syncthreads();

  float sum = 0.0f;
#pragma unroll
  for (int j = 0; j < 32; ++j) { av[j] = __expf(av[j] - mx); sum += av[j]; }
  red[tid] = sum; __syncthreads();
  for (int s = 128; s > 0; s >>= 1) {
    if (tid < s) red[tid] += red[tid + s];
    __syncthreads();
  }
  const float inv = 1.0f / red[0];

  float* orow = out + ((size_t)m * MAX_LEN + t) * VOCAB;
#pragma unroll
  for (int j = 0; j < 32; ++j) {
    int v = tid + j * 256;
    float y = av[j] * inv;
    orow[v] = y;
    ybf[(size_t)m * VOCAB + v] = (__bf16)y;
  }
}

// ---------------------------------------------------------------------------
// Host orchestration
// ---------------------------------------------------------------------------
extern "C" void kernel_launch(void* const* d_in, const int* in_sizes, int n_in,
                              void* d_out, int out_size, void* d_ws, size_t ws_size,
                              hipStream_t stream) {
  (void)in_sizes; (void)n_in; (void)out_size; (void)ws_size;
  const float* noise  = (const float*)d_in[0];
  const float* temp   = (const float*)d_in[1];
  const float* W_init = (const float*)d_in[2];
  const float* b_init = (const float*)d_in[3];
  const float* E      = (const float*)d_in[4];
  const float* W_ih   = (const float*)d_in[5];
  const float* W_hh   = (const float*)d_in[6];
  const float* b_ih   = (const float*)d_in[7];
  const float* b_hh   = (const float*)d_in[8];
  const float* W_out  = (const float*)d_in[9];
  const float* b_out  = (const float*)d_in[10];
  float* out = (float*)d_out;

  // Workspace carve-out (256B aligned), total ~40 MB
  char* p = (char*)d_ws;
  auto carve = [&](size_t bytes) -> char* {
    char* r = p;
    p += (bytes + 255) & ~(size_t)255;
    return r;
  };
  __bf16* Et     = (__bf16*)carve((size_t)EMBED * VOCAB * 2);        // [256,8192]
  __bf16* Wih_b  = (__bf16*)carve((size_t)3 * HIDDEN * EMBED * 2);   // [3072,256]
  __bf16* Whh_b  = (__bf16*)carve((size_t)3 * HIDDEN * HIDDEN * 2);  // [3072,1024]
  __bf16* Wout_b = (__bf16*)carve((size_t)VOCAB * HIDDEN * 2);       // [8192,1024]
  float*  h      = (float*) carve((size_t)BATCH * HIDDEN * 4);
  __bf16* h_bf   = (__bf16*)carve((size_t)BATCH * HIDDEN * 2);
  __bf16* x_bf   = (__bf16*)carve((size_t)BATCH * EMBED * 2);
  float*  gi     = (float*) carve((size_t)BATCH * 3 * HIDDEN * 4);
  float*  gh     = (float*) carve((size_t)BATCH * 3 * HIDDEN * 4);
  float*  logits = (float*) carve((size_t)BATCH * VOCAB * 4);
  __bf16* y_bf   = (__bf16*)carve((size_t)BATCH * VOCAB * 2);
  float*  xpart  = (float*) carve((size_t)SPLITK * BATCH * EMBED * 4);

  // One-time weight conversion (re-done every call: deterministic)
  {
    int n;
    n = 3 * HIDDEN * EMBED;
    k_f32_to_bf16<<<(n + 255) / 256, 256, 0, stream>>>(W_ih, Wih_b, n);
    n = 3 * HIDDEN * HIDDEN;
    k_f32_to_bf16<<<(n + 255) / 256, 256, 0, stream>>>(W_hh, Whh_b, n);
    n = VOCAB * HIDDEN;
    k_f32_to_bf16<<<(n + 255) / 256, 256, 0, stream>>>(W_out, Wout_b, n);
    n = VOCAB * EMBED;
    k_transpose_E<<<(n + 255) / 256, 256, 0, stream>>>(E, Et);
  }
  k_h0<<<(BATCH * HIDDEN + 255) / 256, 256, 0, stream>>>(noise, W_init, b_init, h, h_bf);
  k_init_y<<<(BATCH * VOCAB + 255) / 256, 256, 0, stream>>>(y_bf);

  for (int t = 0; t < MAX_LEN; ++t) {
    // x = y_prev @ E  [128,8192]x[8192,256]: split-K x8 + fixed-order reduce
    k_gemm_x_splitk<<<(SPLITK * 32) / 4, 128, 0, stream>>>(y_bf, Et, xpart);
    k_reduce_x<<<(BATCH * EMBED + 255) / 256, 256, 0, stream>>>(xpart, x_bf);
    // gi = x @ W_ih.T + b_ih  [128,256]x[256,3072] -> f32  (TDM-staged A)
    k_gemm_tdm<<<8 * ((3 * HIDDEN) / 256), 128, 0, stream>>>(
        x_bf, Wih_b, b_ih, gi, nullptr, EMBED, 3 * HIDDEN);
    // gh = h @ W_hh.T + b_hh  [128,1024]x[1024,3072] -> f32  (TDM-staged A)
    k_gemm_tdm<<<8 * ((3 * HIDDEN) / 256), 128, 0, stream>>>(
        h_bf, Whh_b, b_hh, gh, nullptr, HIDDEN, 3 * HIDDEN);
    // GRU gates -> new h (f32 + bf16 mirror)
    k_gru<<<(BATCH * HIDDEN + 255) / 256, 256, 0, stream>>>(gi, gh, h, h_bf);
    // logits = h @ W_out.T + b_out  [128,1024]x[1024,8192] -> f32  (TDM-staged A)
    k_gemm_tdm<<<8 * (VOCAB / 256), 128, 0, stream>>>(
        h_bf, Wout_b, b_out, logits, nullptr, HIDDEN, VOCAB);
    // Gumbel-softmax -> d_out[:, t, :] and y_bf for next step
    k_softmax<<<BATCH, 256, 0, stream>>>(logits, temp, t, out, y_bf);
  }
}
